// RelSAGEConv_11897059410189
// MI455X (gfx1250) — matrix operand order repaired and checked by reading, not compiled, not run on hardware
//
#include <hip/hip_runtime.h>

#define IN_DIM  128
#define EDGE_DIM 64
#define OUT_DIM 128

typedef float    v8f   __attribute__((ext_vector_type(8)));
typedef __bf16   v16bf __attribute__((ext_vector_type(16)));
typedef unsigned v4u   __attribute__((ext_vector_type(4)));

union AFrag {
    unsigned short u16[16];
    v16bf b;
};
union BFrag {
    v4u q[2];
    v16bf b;
};

// fp32 -> bf16 round-to-nearest-even
__device__ __forceinline__ unsigned short f2bf(float f) {
    unsigned u = __float_as_uint(f);
    u += 0x7FFFu + ((u >> 16) & 1u);
    return (unsigned short)(u >> 16);
}

// Opaque value: empty volatile asm so LICM cannot hoist anything whose
// address depends on it out of the tile loop (prevents the compiler from
// promoting 48 LDS-resident weight fragments into VGPRs and spilling them
// to scratch). Emits no instructions; keeps LDS addrspace inference intact
// so fragment reads still lower to ds_load_b128.
__device__ __forceinline__ int opaque(int x) {
    asm volatile("" : "+v"(x));
    return x;
}

// ---------------------------------------------------------------------------
// Tiny precompute: Wc = W_edge @ W_ne[IN_DIM:]   (64 x 128)
// ---------------------------------------------------------------------------
__global__ void wc_kernel(const float* __restrict__ W_edge,
                          const float* __restrict__ W_ne,
                          float* __restrict__ Wc) {
    int idx = blockIdx.x * blockDim.x + threadIdx.x;
    if (idx >= EDGE_DIM * OUT_DIM) return;
    int k = idx >> 7;       // row of W_edge
    int j = idx & 127;      // output column
    float s = 0.0f;
    for (int i = 0; i < OUT_DIM; ++i)
        s += W_edge[k * OUT_DIM + i] * W_ne[(IN_DIM + i) * OUT_DIM + j];
    Wc[idx] = s;
}

// ---------------------------------------------------------------------------
// Edge-message kernel: per 16-edge tile per wave,
//   m = relu( src_h @ W1  +  efeat @ Wc  + b_ne )     (bf16 WMMA, fp32 accum)
// then atomic scatter-add into accumOut[dst] and count into cnt[dst].
// ---------------------------------------------------------------------------
__global__ __launch_bounds__(128, 4) void edge_kernel(
    const float* __restrict__ nfeat, const float* __restrict__ efeat,
    const int*   __restrict__ src,   const int*   __restrict__ dst,
    const float* __restrict__ W_ne,  const float* __restrict__ Wc,
    const float* __restrict__ b_ne,
    float* __restrict__ accumOut, float* __restrict__ cnt,
    int nEdges, int nTiles)
{
    // B fragments, fragment-major: frag f -> 32 lanes x 16 bf16 (32B/lane)
    __shared__ __align__(16) unsigned short ldsW1[32 * 512]; // 4 kb x 8 cb
    __shared__ __align__(16) unsigned short ldsWc[16 * 512]; // 2 kb x 8 cb
    __shared__ float ldsB[OUT_DIM];

    const int tid = threadIdx.x;

    // Stage W1 = W_ne rows [0,128) in WMMA B layout (bf16)
    for (int i = tid; i < 32 * 32; i += 128) {
        int f = i >> 5, l = i & 31;
        int kb = f >> 3, cb = f & 7;
        int col   = cb * 16 + (l & 15);
        int kbase = kb * 32 + (l >> 4) * 8;
        unsigned short* p = &ldsW1[i * 16];
#pragma unroll
        for (int j = 0; j < 16; ++j)
            p[j] = f2bf(W_ne[(kbase + j + (j >= 8 ? 8 : 0)) * OUT_DIM + col]);
    }
    // Stage Wc (64 x 128) in WMMA B layout (bf16)
    for (int i = tid; i < 16 * 32; i += 128) {
        int f = i >> 5, l = i & 31;
        int kb = f >> 3, cb = f & 7;
        int col   = cb * 16 + (l & 15);
        int kbase = kb * 32 + (l >> 4) * 8;
        unsigned short* p = &ldsWc[i * 16];
#pragma unroll
        for (int j = 0; j < 16; ++j)
            p[j] = f2bf(Wc[(kbase + j + (j >= 8 ? 8 : 0)) * OUT_DIM + col]);
    }
    if (tid < OUT_DIM) ldsB[tid] = b_ne[tid];
    __syncthreads();

    const int wave = tid >> 5;
    const int lane = tid & 31;
    const int lhi  = lane >> 4;   // 0/1: which K-half of the A layout
    const int lmod = lane & 15;   // row within tile for A

    for (int tile = blockIdx.x * 4 + wave; tile < nTiles; tile += gridDim.x * 4) {
        const int ebase = tile * 16;
        int myEdge = ebase + lmod;
        if (myEdge >= nEdges) myEdge = nEdges - 1;   // clamp (guards below)
        const float* nrow = nfeat + (size_t)src[myEdge] * IN_DIM;
        const float* erow = efeat + (size_t)myEdge * EDGE_DIM;

        v8f acc[8] = {};   // 8 column-blocks of 16 (fp32 C/D)

        // src_h @ W1 : K = 128 -> 4 K-blocks of 32
#pragma unroll
        for (int kb = 0; kb < 4; ++kb) {
            AFrag a;
            const int b0 = kb * 32 + lhi * 8;
            __align__(16) float t[16];
            *(float4*)&t[0]  = *(const float4*)(nrow + b0);
            *(float4*)&t[4]  = *(const float4*)(nrow + b0 + 4);
            *(float4*)&t[8]  = *(const float4*)(nrow + b0 + 16);
            *(float4*)&t[12] = *(const float4*)(nrow + b0 + 20);
#pragma unroll
            for (int j = 0; j < 16; ++j) a.u16[j] = f2bf(t[j]);

            const int fbase = opaque(kb * 8);   // not hoistable out of tile loop
#pragma unroll
            for (int cb = 0; cb < 8; ++cb) {
                BFrag bw;
                const unsigned short* wp = &ldsW1[((fbase + cb) * 32 + lane) * 16];
                bw.q[0] = *(const v4u*)(wp);
                bw.q[1] = *(const v4u*)(wp + 8);
                acc[cb] = __builtin_amdgcn_wmma_f32_16x16x32_bf16(
                    false, a.b, false, bw.b, (short)0, acc[cb], false, false);
            }
        }
        // efeat @ Wc : K = 64 -> 2 K-blocks of 32
#pragma unroll
        for (int kb = 0; kb < 2; ++kb) {
            AFrag a;
            const int b0 = kb * 32 + lhi * 8;
            __align__(16) float t[16];
            *(float4*)&t[0]  = *(const float4*)(erow + b0);
            *(float4*)&t[4]  = *(const float4*)(erow + b0 + 4);
            *(float4*)&t[8]  = *(const float4*)(erow + b0 + 16);
            *(float4*)&t[12] = *(const float4*)(erow + b0 + 20);
#pragma unroll
            for (int j = 0; j < 16; ++j) a.u16[j] = f2bf(t[j]);

            const int fbase = opaque(kb * 8);
#pragma unroll
            for (int cb = 0; cb < 8; ++cb) {
                BFrag bw;
                const unsigned short* wp = &ldsWc[((fbase + cb) * 32 + lane) * 16];
                bw.q[0] = *(const v4u*)(wp);
                bw.q[1] = *(const v4u*)(wp + 8);
                acc[cb] = __builtin_amdgcn_wmma_f32_16x16x32_bf16(
                    false, a.b, false, bw.b, (short)0, acc[cb], false, false);
            }
        }

        // bias + relu + atomic scatter (segment sum).
        // D layout: VGPR r, lane l -> M = r + 8*(l>=16), N = l%16
#pragma unroll
        for (int r = 0; r < 8; ++r) {
            const int e = ebase + r + lhi * 8;
            if (e < nEdges) {
                float* drow = accumOut + (size_t)dst[e] * OUT_DIM;
#pragma unroll
                for (int cb = 0; cb < 8; ++cb) {
                    const int col = cb * 16 + lmod;
                    float v = acc[cb][r] + ldsB[col];
                    v = v > 0.0f ? v : 0.0f;
                    atomicAdd(drow + col, v);
                }
            }
        }
        if (lane < 16 && (ebase + lane) < nEdges)
            atomicAdd(cnt + dst[ebase + lane], 1.0f);
    }
}

// ---------------------------------------------------------------------------
// Node kernel: out = summ/max(cnt,1) + nfeat @ W_self + b_self
// (out already holds summ from the edge kernel's atomics)
// ---------------------------------------------------------------------------
__global__ __launch_bounds__(128, 4) void self_kernel(
    const float* __restrict__ nfeat, const float* __restrict__ W_self,
    const float* __restrict__ b_self, const float* __restrict__ cnt,
    float* __restrict__ out, int nNodes, int nTiles)
{
    __shared__ __align__(16) unsigned short ldsW[32 * 512];
    __shared__ float ldsB[OUT_DIM];

    const int tid = threadIdx.x;
    for (int i = tid; i < 32 * 32; i += 128) {
        int f = i >> 5, l = i & 31;
        int kb = f >> 3, cb = f & 7;
        int col   = cb * 16 + (l & 15);
        int kbase = kb * 32 + (l >> 4) * 8;
        unsigned short* p = &ldsW[i * 16];
#pragma unroll
        for (int j = 0; j < 16; ++j)
            p[j] = f2bf(W_self[(kbase + j + (j >= 8 ? 8 : 0)) * OUT_DIM + col]);
    }
    if (tid < OUT_DIM) ldsB[tid] = b_self[tid];
    __syncthreads();

    const int wave = tid >> 5;
    const int lane = tid & 31;
    const int lhi  = lane >> 4;
    const int lmod = lane & 15;

    for (int tile = blockIdx.x * 4 + wave; tile < nTiles; tile += gridDim.x * 4) {
        const int nbase = tile * 16;
        int myNode = nbase + lmod;
        if (myNode >= nNodes) myNode = nNodes - 1;
        const float* nrow = nfeat + (size_t)myNode * IN_DIM;

        v8f acc[8] = {};
#pragma unroll
        for (int kb = 0; kb < 4; ++kb) {
            AFrag a;
            const int b0 = kb * 32 + lhi * 8;
            __align__(16) float t[16];
            *(float4*)&t[0]  = *(const float4*)(nrow + b0);
            *(float4*)&t[4]  = *(const float4*)(nrow + b0 + 4);
            *(float4*)&t[8]  = *(const float4*)(nrow + b0 + 16);
            *(float4*)&t[12] = *(const float4*)(nrow + b0 + 20);
#pragma unroll
            for (int j = 0; j < 16; ++j) a.u16[j] = f2bf(t[j]);

            const int fbase = opaque(kb * 8);
#pragma unroll
            for (int cb = 0; cb < 8; ++cb) {
                BFrag bw;
                const unsigned short* wp = &ldsW[((fbase + cb) * 32 + lane) * 16];
                bw.q[0] = *(const v4u*)(wp);
                bw.q[1] = *(const v4u*)(wp + 8);
                acc[cb] = __builtin_amdgcn_wmma_f32_16x16x32_bf16(
                    false, a.b, false, bw.b, (short)0, acc[cb], false, false);
            }
        }

#pragma unroll
        for (int r = 0; r < 8; ++r) {
            const int node = nbase + r + lhi * 8;
            if (node < nNodes) {
                float c = cnt[node];
                float inv = 1.0f / (c > 1.0f ? c : 1.0f);
                float* orow = out + (size_t)node * OUT_DIM;
#pragma unroll
                for (int cb = 0; cb < 8; ++cb) {
                    const int col = cb * 16 + lmod;
                    orow[col] = orow[col] * inv + acc[cb][r] + ldsB[col];
                }
            }
        }
    }
}

// ---------------------------------------------------------------------------
extern "C" void kernel_launch(void* const* d_in, const int* in_sizes, int n_in,
                              void* d_out, int out_size, void* d_ws, size_t ws_size,
                              hipStream_t stream) {
    const float* nfeat  = (const float*)d_in[0];
    const float* efeat  = (const float*)d_in[1];
    const int*   src    = (const int*)d_in[2];
    const int*   dst    = (const int*)d_in[3];
    const float* W_edge = (const float*)d_in[4];
    const float* W_ne   = (const float*)d_in[5];
    const float* b_ne   = (const float*)d_in[6];
    const float* W_self = (const float*)d_in[7];
    const float* b_self = (const float*)d_in[8];

    const int nEdges = in_sizes[2];
    const int nNodes = in_sizes[0] / IN_DIM;

    float* Wc  = (float*)d_ws;                                               // 64*128 f32
    float* cnt = (float*)((char*)d_ws + EDGE_DIM * OUT_DIM * sizeof(float)); // nNodes f32
    float* out = (float*)d_out;

    // Zero accumulators (graph-capture safe async memsets)
    hipMemsetAsync(d_out, 0, (size_t)out_size * sizeof(float), stream);
    hipMemsetAsync(cnt, 0, (size_t)nNodes * sizeof(float), stream);

    // Wc = W_edge @ W_ne[IN_DIM:]
    wc_kernel<<<(EDGE_DIM * OUT_DIM + 127) / 128, 128, 0, stream>>>(W_edge, W_ne, Wc);

    // Edge messages + scatter
    const int nTilesE = (nEdges + 15) / 16;
    int gridE = (nTilesE + 3) / 4;
    if (gridE > 1024) gridE = 1024;
    edge_kernel<<<gridE, 128, 0, stream>>>(nfeat, efeat, src, dst, W_ne, Wc, b_ne,
                                           out, cnt, nEdges, nTilesE);

    // Mean + self transform + bias
    const int nTilesN = (nNodes + 15) / 16;
    int gridN = (nTilesN + 3) / 4;
    if (gridN > 1024) gridN = 1024;
    self_kernel<<<gridN, 128, 0, stream>>>(nfeat, W_self, b_self, cnt,
                                           out, nNodes, nTilesN);
}
